// MultiHeadSelfAttention_7112465842185
// MI455X (gfx1250) — matrix operand order repaired and checked by reading, not compiled, not run on hardware
//
#include <hip/hip_runtime.h>
#include <hip/hip_bf16.h>

typedef __attribute__((ext_vector_type(16))) __bf16 v16bf;
typedef __attribute__((ext_vector_type(8)))  float  v8f;
typedef __attribute__((ext_vector_type(4)))  float  f4;

#define NUM_HEADS 16
#define SEQ 256
#define DH 64
#define CIN 3072
#define QK_STRIDE 72      // 64 + 8 pad (144B rows: conflict-free, 8B aligned)
#define VT_STRIDE 264     // 256 + 8 pad
#define PST_STRIDE 40     // 32 + 8 pad
#define NTHREADS 512
#define LOG2E 1.4426950408889634f

__device__ __forceinline__ unsigned short f2bf(float f) {
  unsigned int u = __float_as_uint(f);
  unsigned int r = u + 0x7FFFu + ((u >> 16) & 1u);   // round-to-nearest-even
  return (unsigned short)(r >> 16);
}
__device__ __forceinline__ unsigned int pack2bf(float a, float b) {
  return (unsigned int)f2bf(a) | ((unsigned int)f2bf(b) << 16);
}

// Butterfly reduction within 16-lane halves via single ds_swizzle_b32 ops
// (group-of-32 mode: and=0x1F, or=0, xor=M).
template <int M>
__device__ __forceinline__ float swz_max(float v) {
  float o = __int_as_float(__builtin_amdgcn_ds_swizzle(__float_as_int(v), 0x7C00 | M));
  return fmaxf(v, o);
}
template <int M>
__device__ __forceinline__ float swz_add(float v) {
  float o = __int_as_float(__builtin_amdgcn_ds_swizzle(__float_as_int(v), 0x7C00 | M));
  return v + o;
}

union FragU { v16bf v; uint2 q[4]; };

// A-matrix 16x32 bf16 fragment (ISA 7.12.2): lane l -> row r0+(l&15),
// k = k0 + (l>>4)*8 + {0..7} in v0..3 and k0 + 16 + (l>>4)*8 + {0..7} in v4..7.
__device__ __forceinline__ v16bf load_fragA(const unsigned short* mat, int stride,
                                            int r0, int k0, int lane) {
  int row  = r0 + (lane & 15);
  int base = k0 + ((lane >> 4) << 3);
  const unsigned short* p = mat + row * stride + base;
  FragU f;
  f.q[0] = *(const uint2*)(p + 0);
  f.q[1] = *(const uint2*)(p + 4);
  f.q[2] = *(const uint2*)(p + 16);
  f.q[3] = *(const uint2*)(p + 20);
  return f.v;
}

// B-matrix 32x16 bf16 fragment: lane l -> column (l&15),
// k = k0 + (l>>4)*8 + {0..7} in v0..3 and k0 + 16 + (l>>4)*8 + {0..7} in v4..7.
__device__ __forceinline__ v16bf load_fragB(const unsigned short* mat, int stride,
                                            int r0, int k0, int lane) {
  int row  = r0 + (lane & 15);
  int base = k0 + ((lane >> 4) << 3);
  const unsigned short* p = mat + row * stride + base;
  FragU f;
  f.q[0] = *(const uint2*)(p + 0);
  f.q[1] = *(const uint2*)(p + 4);
  f.q[2] = *(const uint2*)(p + 16);
  f.q[3] = *(const uint2*)(p + 20);
  return f.v;
}

__device__ __forceinline__ v8f wmma_bf16(v16bf a, v16bf b, v8f c) {
  return __builtin_amdgcn_wmma_f32_16x16x32_bf16(false, a, false, b, (short)0, c,
                                                 false, false);
}

__global__ __launch_bounds__(NTHREADS)
void mhsa_wmma_kernel(const float* __restrict__ x,
                      const float* __restrict__ bias_table,
                      float* __restrict__ out) {
  extern __shared__ unsigned short smem[];
  unsigned short* Qs  = smem;                              // 256*72 bf16
  unsigned short* Ks  = Qs + SEQ * QK_STRIDE;              // 256*72 bf16
  unsigned short* Vt  = Ks + SEQ * QK_STRIDE;              // 64*264 bf16 (V transposed)
  unsigned short* Pst = Vt + DH * VT_STRIDE;               // 16 waves * 16*40 bf16
  float* lbias = (float*)(Pst + 16 * 16 * PST_STRIDE);     // 511 floats (bias*log2e)

  const int bh   = blockIdx.x;
  const int b    = bh >> 4;
  const int h    = bh & (NUM_HEADS - 1);
  const int tid  = threadIdx.x;
  const int lane = tid & 31;
  const int wave = tid >> 5;

  const float* xb = x + (size_t)b * SEQ * CIN + h * DH;

  // Bias column for this head, pre-scaled into log2 domain.
  for (int i = tid; i < 2 * SEQ - 1; i += NTHREADS)
    lbias[i] = bias_table[i * NUM_HEADS + h] * LOG2E;

  // Single streaming pass over x: fp32 -> bf16 into LDS. Q/K row-major, V transposed.
  // Non-temporal: every byte of x is read exactly once grid-wide.
  for (int i = tid; i < SEQ * 16; i += NTHREADS) {
    int row = i >> 4;
    int c4  = (i & 15) << 2;
    const float* src = xb + (size_t)row * CIN + c4;
    f4 q = __builtin_nontemporal_load((const f4*)(src));
    f4 k = __builtin_nontemporal_load((const f4*)(src + 1024));
    f4 v = __builtin_nontemporal_load((const f4*)(src + 2048));
    uint2 pq; pq.x = pack2bf(q.x, q.y); pq.y = pack2bf(q.z, q.w);
    uint2 pk; pk.x = pack2bf(k.x, k.y); pk.y = pack2bf(k.z, k.w);
    *(uint2*)&Qs[row * QK_STRIDE + c4] = pq;
    *(uint2*)&Ks[row * QK_STRIDE + c4] = pk;
    Vt[(c4 + 0) * VT_STRIDE + row] = f2bf(v.x);
    Vt[(c4 + 1) * VT_STRIDE + row] = f2bf(v.y);
    Vt[(c4 + 2) * VT_STRIDE + row] = f2bf(v.z);
    Vt[(c4 + 3) * VT_STRIDE + row] = f2bf(v.w);
  }
  __syncthreads();

  const int m0 = wave << 4;            // this wave's 16 query rows
  const int lo = lane & 15;
  const int hi = lane >> 4;
  // C-tile element (vgpr r, lane): M = r + 8*hi, N = lo; bias idx = m - n + 255.
  const int bias_base = m0 + (hi << 3) - lo + (SEQ - 1);

  v16bf aQ0 = load_fragA(Qs, QK_STRIDE, m0, 0,  lane);
  v16bf aQ1 = load_fragA(Qs, QK_STRIDE, m0, 32, lane);

  unsigned short* Pw = Pst + wave * 16 * PST_STRIDE;
  v8f acc[4];
  float rmax[8], rsum[8];
#pragma unroll
  for (int r = 0; r < 8; ++r) { rmax[r] = -3.0e38f; rsum[r] = 0.f; }
#pragma unroll
  for (int j = 0; j < 4; ++j)
#pragma unroll
    for (int r = 0; r < 8; ++r) acc[j][r] = 0.f;

  // ---- fused online-softmax attention over 8 key blocks of 32 (log2 domain) ----
#pragma unroll
  for (int s = 0; s < 8; ++s) {
    const int t0 = 2 * s, t1 = 2 * s + 1;

    // GEMM1: two 16x16 score tiles for keys [32s, 32s+32)
    v16bf bK0 = load_fragB(Ks, QK_STRIDE, t0 << 4, 0,  lane);
    v16bf bK1 = load_fragB(Ks, QK_STRIDE, t0 << 4, 32, lane);
    v8f c0 = {0.f, 0.f, 0.f, 0.f, 0.f, 0.f, 0.f, 0.f};
    c0 = wmma_bf16(aQ0, bK0, c0);
    c0 = wmma_bf16(aQ1, bK1, c0);
    bK0 = load_fragB(Ks, QK_STRIDE, t1 << 4, 0,  lane);
    bK1 = load_fragB(Ks, QK_STRIDE, t1 << 4, 32, lane);
    v8f c1 = {0.f, 0.f, 0.f, 0.f, 0.f, 0.f, 0.f, 0.f};
    c1 = wmma_bf16(aQ0, bK0, c1);
    c1 = wmma_bf16(aQ1, bK1, c1);

    // scale (0.25 * log2e) + relative-position bias (already * log2e)
    const int bb0 = bias_base - (t0 << 4);
    const int bb1 = bias_base - (t1 << 4);
#pragma unroll
    for (int r = 0; r < 8; ++r) {
      c0[r] = fmaf(c0[r], 0.25f * LOG2E, lbias[bb0 + r]);
      c1[r] = fmaf(c1[r], 0.25f * LOG2E, lbias[bb1 + r]);
    }

    // new row maxima (ds_swizzle butterfly within 16-lane halves)
    float nm[8];
    bool grew = false;
#pragma unroll
    for (int r = 0; r < 8; ++r) {
      float m = fmaxf(fmaxf(c0[r], c1[r]), rmax[r]);
      m = swz_max<1>(m);
      m = swz_max<2>(m);
      m = swz_max<4>(m);
      m = swz_max<8>(m);
      nm[r] = m;
      grew |= (m > rmax[r]);
    }
    // rescale accumulator/sums only if some row max grew (wave-uniform branch;
    // EXEC stays all-1s so WMMA legality is unaffected)
    if (__ballot(grew)) {
#pragma unroll
      for (int r = 0; r < 8; ++r) {
        float f = __builtin_amdgcn_exp2f(rmax[r] - nm[r]);  // first iter: 0
        rsum[r] *= f;
#pragma unroll
        for (int j = 0; j < 4; ++j) acc[j][r] *= f;
      }
    }
    // exp2, accumulate lane-partial row sums, stage P as bf16
#pragma unroll
    for (int r = 0; r < 8; ++r) {
      rmax[r] = nm[r];
      float p0 = __builtin_amdgcn_exp2f(c0[r] - nm[r]);
      float p1 = __builtin_amdgcn_exp2f(c1[r] - nm[r]);
      rsum[r] += p0 + p1;
      Pw[(r + (hi << 3)) * PST_STRIDE + lo]      = f2bf(p0);
      Pw[(r + (hi << 3)) * PST_STRIDE + 16 + lo] = f2bf(p1);
    }

    // GEMM2 partial: acc += P_block * V_block (same-wave DS ops are in-order)
    v16bf aP = load_fragA(Pw, PST_STRIDE, 0, 0, lane);
#pragma unroll
    for (int j = 0; j < 4; ++j) {              // 4 d-tiles (Dh = 64)
      v16bf bV = load_fragB(Vt, VT_STRIDE, j << 4, s << 5, lane);
      acc[j] = wmma_bf16(aP, bV, acc[j]);
    }
  }

  // ---- final normalization: reduce lane-partial row sums ----
#pragma unroll
  for (int r = 0; r < 8; ++r) {
    float sum = rsum[r];
    sum = swz_add<1>(sum);
    sum = swz_add<2>(sum);
    sum = swz_add<4>(sum);
    sum = swz_add<8>(sum);
    rsum[r] = __builtin_amdgcn_rcpf(sum);
  }

  // ---- store out[b, m, h*64 + d], normalized (non-temporal stream-out) ----
  float* ob = out + ((size_t)b * SEQ + m0) * (NUM_HEADS * DH) + h * DH;
#pragma unroll
  for (int j = 0; j < 4; ++j)
#pragma unroll
    for (int r = 0; r < 8; ++r) {
      int m = r + (hi << 3);
      __builtin_nontemporal_store(acc[j][r] * rsum[r],
                                  &ob[(size_t)m * (NUM_HEADS * DH) + (j << 4) + lo]);
    }
}

extern "C" void kernel_launch(void* const* d_in, const int* in_sizes, int n_in,
                              void* d_out, int out_size, void* d_ws, size_t ws_size,
                              hipStream_t stream) {
  const float* x          = (const float*)d_in[0];
  const float* bias_table = (const float*)d_in[1];
  float* out              = (float*)d_out;

  const size_t shmem =
      (size_t)(2 * SEQ * QK_STRIDE + DH * VT_STRIDE + 16 * 16 * PST_STRIDE) *
          sizeof(unsigned short) +
      (size_t)(2 * SEQ) * sizeof(float);   // 130048 bytes; 2 workgroups / 320KB WGP

  hipFuncSetAttribute((const void*)mhsa_wmma_kernel,
                      hipFuncAttributeMaxDynamicSharedMemorySize, (int)shmem);

  mhsa_wmma_kernel<<<dim3(128 * NUM_HEADS), dim3(NTHREADS), shmem, stream>>>(
      x, bias_table, out);
}
